// PhysicsRRN_18296560681665
// MI455X (gfx1250) — matrix operand order repaired and checked by reading, not compiled, not run on hardware
//
#include <hip/hip_runtime.h>
#include <hip/hip_bf16.h>

typedef __attribute__((ext_vector_type(16))) _Float16 v16h;
typedef __attribute__((ext_vector_type(8)))  float    v8f;

#define BSZ    1024
#define NNODE  32
#define TDIM   128
#define FDIM   4
#define HDIM   16
#define STEPS  8
#define EDGES  992      // 32*31 directed edges per graph
#define ETILES 62       // 992 / 16 — exact
#define EPAIRS 31       // tiles processed two-at-a-time per wave (ILP)
#define WAVES  4
#define BLOCK  (WAVES * 32)

// Build a B-matrix fragment (32x16, f16) for V_WMMA_F32_16X16X32_F16 from a
// row-major f32 weight matrix W[din][dout], zero-padded to K=32 / N=16.
// Layout: lane L holds column n = L%16; h<8 holds K = 8*(L/16) + h, h>=8 holds K = 16 + 8*(L/16) + (h-8).
__device__ inline v16h build_B(const float* __restrict__ W, int din, int dout, int lane) {
    v16h B;
    const int n  = lane & 15;
    const int kb = (lane >> 4) * 16;
#pragma unroll
    for (int h = 0; h < 16; ++h) {
        const int k = (h < 8) ? (kb >> 1) + h : 16 + (kb >> 1) + (h - 8);
        float v = 0.0f;
        if (k < din && n < dout) v = W[k * dout + n];
        B[h] = (_Float16)v;
    }
    return B;
}

__device__ inline v8f wmma_f16(v16h a, v16h b, v8f c) {
    return __builtin_amdgcn_wmma_f32_16x16x32_f16(false, a, false, b, (short)0, c, false, false);
}

// edge e in [0, 992): src i = e/31 ; dst j = (e%31) + (e%31 >= i)
__device__ inline void edge_ij(int e, int& i, int& j) {
    i = e / 31;
    const int jj = e - i * 31;
    j = jj + (jj >= i ? 1 : 0);
}

__global__ void __launch_bounds__(BLOCK)
rrn_kernel(const float* __restrict__ inputs,
           const long long* __restrict__ edges,   // structure is implicit; unused
           const float* mW1, const float* mb1, const float* mW2, const float* mb2,
           const float* mW3, const float* mb3, const float* mW4, const float* mb4,
           const float* oW1, const float* ob1, const float* oW2, const float* ob2,
           const float* oW3, const float* ob3, const float* oW4, const float* ob4,
           float* __restrict__ out) {
    (void)edges;
    const int g    = blockIdx.x;        // graph id
    const int tid  = threadIdx.x;
    const int lane = tid & 31;          // wave32
    const int wave = tid >> 5;
    const int n16  = lane & 15;         // N column / activation row owned by this lane
    const int hi   = lane >> 4;         // half-wave select

    __shared__ float x_sh[NNODE][FDIM];          // node states
    __shared__ float acc[NNODE][FDIM];           // segment-sum accumulator
    __shared__ float hbuf[WAVES][2][16][17];     // per-wave, per-chain transpose buffers

    // ---- resident weight fragments (per wave, in VGPRs) ----
    const v16h Bm1 = build_B(mW1, 2 * FDIM + 1, HDIM, lane);   // 9  -> 16
    const v16h Bm2 = build_B(mW2, HDIM, HDIM, lane);           // 16 -> 16
    const v16h Bm3 = build_B(mW3, HDIM, HDIM, lane);           // 16 -> 16
    const v16h Bm4 = build_B(mW4, HDIM, FDIM, lane);           // 16 -> 4 (N-padded)
    const v16h Bo1 = build_B(oW1, FDIM, HDIM, lane);           // 4  -> 16
    const v16h Bo2 = build_B(oW2, HDIM, HDIM, lane);
    const v16h Bo3 = build_B(oW3, HDIM, HDIM, lane);
    const v16h Bo4 = build_B(oW4, HDIM, 2, lane);              // 16 -> 2 (N-padded)

    // per-lane bias scalars (each lane owns output column n16)
    const float bm1v = mb1[n16];
    const float bm2v = mb2[n16];
    const float bm3v = mb3[n16];
    const float bm4v = (n16 < FDIM) ? mb4[n16] : 0.0f;
    const float bo1v = ob1[n16];
    const float bo2v = ob2[n16];
    const float bo3v = ob3[n16];
    const float bo4v = (n16 < 2) ? ob4[n16] : 0.0f;

    // ---- init: x = inputs[g, :, 0, :]; acc = 0 ----
    for (int idx = tid; idx < NNODE * FDIM; idx += BLOCK) {
        const int n = idx >> 2, f = idx & 3;
        x_sh[n][f] = inputs[(((size_t)g * NNODE + n) * TDIM + 0) * FDIM + f];
        acc[n][f]  = 0.0f;
    }
    __syncthreads();

    for (int step = 0; step < STEPS; ++step) {
        // ========== message MLP: 31 tile-pairs, 2 independent chains per wave ==========
        for (int tp = wave; tp < EPAIRS; tp += WAVES) {
            const int tile[2] = { 2 * tp, 2 * tp + 1 };

            // ---- layer 1 A: [x[src](4) | x[dst](4) | ef=0 | pad] ----
            // lanes 0-15 carry K=0..7; lanes 16-31 carry K=8..15 (all zero: ef & pad)
            v16h A[2] = { {}, {} };
            if (hi == 0) {
#pragma unroll
                for (int u = 0; u < 2; ++u) {
                    int i, j;
                    edge_ij(tile[u] * 16 + n16, i, j);
                    A[u][0] = (_Float16)x_sh[i][0];
                    A[u][1] = (_Float16)x_sh[i][1];
                    A[u][2] = (_Float16)x_sh[i][2];
                    A[u][3] = (_Float16)x_sh[i][3];
                    A[u][4] = (_Float16)x_sh[j][0];
                    A[u][5] = (_Float16)x_sh[j][1];
                    A[u][6] = (_Float16)x_sh[j][2];
                    A[u][7] = (_Float16)x_sh[j][3];
                }
            }
            v8f C[2];
#pragma unroll
            for (int u = 0; u < 2; ++u) { v8f z = {}; C[u] = wmma_f16(A[u], Bm1, z); }

            // ---- layers 2..4: bias+relu, LDS transpose, WMMA (chains interleaved) ----
#pragma unroll
            for (int u = 0; u < 2; ++u)
#pragma unroll
                for (int v = 0; v < 8; ++v)
                    hbuf[wave][u][v + 8 * hi][n16] = fmaxf(C[u][v] + bm1v, 0.0f);
#pragma unroll
            for (int u = 0; u < 2; ++u) {
                v16h A2 = {};
#pragma unroll
                for (int h = 0; h < 8; ++h)
                    A2[h] = (_Float16)hbuf[wave][u][n16][8 * hi + h];
                A[u] = A2;
            }
#pragma unroll
            for (int u = 0; u < 2; ++u) { v8f z = {}; C[u] = wmma_f16(A[u], Bm2, z); }

#pragma unroll
            for (int u = 0; u < 2; ++u)
#pragma unroll
                for (int v = 0; v < 8; ++v)
                    hbuf[wave][u][v + 8 * hi][n16] = fmaxf(C[u][v] + bm2v, 0.0f);
#pragma unroll
            for (int u = 0; u < 2; ++u) {
                v16h A3 = {};
#pragma unroll
                for (int h = 0; h < 8; ++h)
                    A3[h] = (_Float16)hbuf[wave][u][n16][8 * hi + h];
                A[u] = A3;
            }
#pragma unroll
            for (int u = 0; u < 2; ++u) { v8f z = {}; C[u] = wmma_f16(A[u], Bm3, z); }

#pragma unroll
            for (int u = 0; u < 2; ++u)
#pragma unroll
                for (int v = 0; v < 8; ++v)
                    hbuf[wave][u][v + 8 * hi][n16] = fmaxf(C[u][v] + bm3v, 0.0f);
#pragma unroll
            for (int u = 0; u < 2; ++u) {
                v16h A4 = {};
#pragma unroll
                for (int h = 0; h < 8; ++h)
                    A4[h] = (_Float16)hbuf[wave][u][n16][8 * hi + h];
                A[u] = A4;
            }
#pragma unroll
            for (int u = 0; u < 2; ++u) { v8f z = {}; C[u] = wmma_f16(A[u], Bm4, z); }

            // ---- segment-sum: lanes with n16<4 hold message feature n16 of 8 edges ----
            if (n16 < FDIM) {
#pragma unroll
                for (int u = 0; u < 2; ++u)
#pragma unroll
                    for (int v = 0; v < 8; ++v) {
                        int i, j;
                        edge_ij(tile[u] * 16 + v + 8 * hi, i, j);
                        atomicAdd(&acc[j][n16], C[u][v] + bm4v);   // ds_add_f32
                    }
            }
        }
        __syncthreads();

        // ---- x = segment_sum result; reset accumulator ----
        for (int idx = tid; idx < NNODE * FDIM; idx += BLOCK) {
            const int n = idx >> 2, f = idx & 3;
            x_sh[n][f] = acc[n][f];
            acc[n][f]  = 0.0f;
        }
        __syncthreads();

        // ========== readout MLP: 32 nodes = 2 tiles (waves 0,1) ==========
        if (wave < 2) {
            const int t = wave;
            v16h A = {};
            if (hi == 0) {           // K=0..3 valid; everything else zero
                const int node = t * 16 + n16;
                A[0] = (_Float16)x_sh[node][0];
                A[1] = (_Float16)x_sh[node][1];
                A[2] = (_Float16)x_sh[node][2];
                A[3] = (_Float16)x_sh[node][3];
            }
            v8f C = {};
            C = wmma_f16(A, Bo1, C);
#pragma unroll
            for (int v = 0; v < 8; ++v)
                hbuf[wave][0][v + 8 * hi][n16] = fmaxf(C[v] + bo1v, 0.0f);
            v16h A2 = {};
#pragma unroll
            for (int h = 0; h < 8; ++h)
                A2[h] = (_Float16)hbuf[wave][0][n16][8 * hi + h];
            v8f C2 = {};
            C2 = wmma_f16(A2, Bo2, C2);
#pragma unroll
            for (int v = 0; v < 8; ++v)
                hbuf[wave][0][v + 8 * hi][n16] = fmaxf(C2[v] + bo2v, 0.0f);
            v16h A3 = {};
#pragma unroll
            for (int h = 0; h < 8; ++h)
                A3[h] = (_Float16)hbuf[wave][0][n16][8 * hi + h];
            v8f C3 = {};
            C3 = wmma_f16(A3, Bo3, C3);
#pragma unroll
            for (int v = 0; v < 8; ++v)
                hbuf[wave][0][v + 8 * hi][n16] = fmaxf(C3[v] + bo3v, 0.0f);
            v16h A4 = {};
#pragma unroll
            for (int h = 0; h < 8; ++h)
                A4[h] = (_Float16)hbuf[wave][0][n16][8 * hi + h];
            v8f C4 = {};
            C4 = wmma_f16(A4, Bo4, C4);

            if (n16 < 2) {
#pragma unroll
                for (int v = 0; v < 8; ++v) {
                    const int node = t * 16 + v + 8 * hi;
                    out[(((size_t)step * BSZ + g) * NNODE + node) * 2 + n16] = C4[v] + bo4v;
                }
            }
        }
        __syncthreads();
    }
}

extern "C" void kernel_launch(void* const* d_in, const int* in_sizes, int n_in,
                              void* d_out, int out_size, void* d_ws, size_t ws_size,
                              hipStream_t stream) {
    (void)in_sizes; (void)n_in; (void)out_size; (void)d_ws; (void)ws_size;
    const float*     inputs = (const float*)d_in[0];
    const long long* edges  = (const long long*)d_in[1];
    const float* mW1 = (const float*)d_in[2];
    const float* mb1 = (const float*)d_in[3];
    const float* mW2 = (const float*)d_in[4];
    const float* mb2 = (const float*)d_in[5];
    const float* mW3 = (const float*)d_in[6];
    const float* mb3 = (const float*)d_in[7];
    const float* mW4 = (const float*)d_in[8];
    const float* mb4 = (const float*)d_in[9];
    const float* oW1 = (const float*)d_in[10];
    const float* ob1 = (const float*)d_in[11];
    const float* oW2 = (const float*)d_in[12];
    const float* ob2 = (const float*)d_in[13];
    const float* oW3 = (const float*)d_in[14];
    const float* ob3 = (const float*)d_in[15];
    const float* oW4 = (const float*)d_in[16];
    const float* ob4 = (const float*)d_in[17];
    float* out = (float*)d_out;

    rrn_kernel<<<BSZ, BLOCK, 0, stream>>>(inputs, edges,
                                          mW1, mb1, mW2, mb2, mW3, mb3, mW4, mb4,
                                          oW1, ob1, oW2, ob2, oW3, ob3, oW4, ob4,
                                          out);
}
